// BidirectionalMamba_2731599200642
// MI455X (gfx1250) — matrix operand order repaired and checked by reading, not compiled, not run on hardware
//
#include <hip/hip_runtime.h>
#include <hip/hip_bf16.h>

#define D_MODEL 256
#define D_INNER 512
#define D_STATE 16
#define D_CONV  4
#define DT_RANK 16
#define BB      2
#define TT      4096
#define MROWS   (BB * TT)   // 8192

typedef __attribute__((ext_vector_type(16))) __bf16 v16bf;
typedef __attribute__((ext_vector_type(8)))  float  v8f;

static __device__ __forceinline__ float silu_f(float v) {
    return v / (1.0f + expf(-v));
}

// ---------------------------------------------------------------------------
// Weight fusion:  W_eff[i,j] = sum_k W_out[i,k] * Msub[k,j]   (512x256, K=256)
// ---------------------------------------------------------------------------
__global__ void fuse_w_kernel(const float* __restrict__ Wout,
                              const float* __restrict__ Msub,
                              float* __restrict__ Weff) {
    int idx = blockIdx.x * blockDim.x + threadIdx.x;
    if (idx >= D_INNER * D_MODEL) return;
    int i = idx / D_MODEL;
    int j = idx - i * D_MODEL;
    float acc = 0.0f;
    const float* wr = Wout + (size_t)i * D_MODEL;
    #pragma unroll 4
    for (int k = 0; k < D_MODEL; ++k)
        acc = fmaf(wr[k], Msub[(size_t)k * D_MODEL + j], acc);
    Weff[idx] = acc;
}

// ---------------------------------------------------------------------------
// Pack a row-major f32 weight (K x N) into bf16 WMMA B-fragment order:
//   dst[((ktile*2 + half)*N + n)*16 + e] = src[(ktile*32 + half*16 + e)*N + n]
// so each lane's 16 B-fragment elements are a contiguous 32-byte run.
// ---------------------------------------------------------------------------
__global__ void pack_b_kernel(const float* __restrict__ in,
                              __bf16* __restrict__ out, int K, int N) {
    int idx = blockIdx.x * blockDim.x + threadIdx.x;
    if (idx >= K * N) return;
    int k = idx / N;
    int n = idx - k * N;
    int kt   = k >> 5;
    int half = (k >> 4) & 1;
    int e    = k & 15;
    out[(((size_t)kt * 2 + half) * N + n) * 16 + e] = (__bf16)in[idx];
}

// ---------------------------------------------------------------------------
// Wave32 WMMA GEMM, 16(M) x (16*NT)(N) tile per wave:
//   C[M,N] (+)= A'[M,K] * B[K,N]
//   A  : f32 row-major (lda), converted to bf16 in-register (4x b128 per frag)
//   Bp : bf16 pre-packed fragment-order (see pack_b_kernel)
//   flipT : 0 -> A' = A ; else rows time-flipped within each batch of flipT
//   accum : 1 -> C += , 0 -> C =
// A/B fragments use contiguous k order, k = half*16 + e, in both operands (wave32).
// ---------------------------------------------------------------------------
template <int NT>
__global__ void wmma_gemm_kernel(const float* __restrict__ A,
                                 const __bf16* __restrict__ Bp,
                                 float* __restrict__ C,
                                 int M, int N, int K,
                                 int lda, int ldc,
                                 int flipT, int accum) {
    const int lane = threadIdx.x & 31;
    const int n16  = lane & 15;
    const int half = lane >> 4;

    const int tn = blockIdx.x * (16 * NT);
    const int tm = blockIdx.y * 16;

    // A-fragment row for this lane (lanes l and l+16 share the row, split K)
    int mrow = tm + n16;
    int arow = mrow;
    if (flipT) {
        int b = mrow / flipT;
        int t = mrow - b * flipT;
        arow  = b * flipT + (flipT - 1 - t);
    }
    const float* __restrict__ Arow = A + (size_t)arow * lda;

    v8f c[NT];
    #pragma unroll
    for (int j = 0; j < NT; ++j) c[j] = (v8f){};
    if (accum) {
        #pragma unroll
        for (int j = 0; j < NT; ++j)
            #pragma unroll
            for (int r = 0; r < 8; ++r)
                c[j][r] = C[(size_t)(tm + r + 8 * half) * ldc + tn + j * 16 + n16];
    }

    for (int k0 = 0; k0 < K; k0 += 32) {
        if (k0 + 32 < K)  // global_prefetch_b8 (speculative, OOB-safe)
            __builtin_prefetch(Arow + k0 + 32, 0, 0);

        // A 16x32 bf16 fragment = two contiguous 8-float runs per lane
        const float* s0 = Arow + k0 + half * 16;       // K = half*16 + 0..7
        const float* s1 = s0 + 8;                      // K = half*16 + 8..15
        float4 a0 = *(const float4*)(s0);
        float4 a1 = *(const float4*)(s0 + 4);
        float4 a2 = *(const float4*)(s1);
        float4 a3 = *(const float4*)(s1 + 4);
        v16bf a;
        a[0]  = (__bf16)a0.x; a[1]  = (__bf16)a0.y; a[2]  = (__bf16)a0.z; a[3]  = (__bf16)a0.w;
        a[4]  = (__bf16)a1.x; a[5]  = (__bf16)a1.y; a[6]  = (__bf16)a1.z; a[7]  = (__bf16)a1.w;
        a[8]  = (__bf16)a2.x; a[9]  = (__bf16)a2.y; a[10] = (__bf16)a2.z; a[11] = (__bf16)a2.w;
        a[12] = (__bf16)a3.x; a[13] = (__bf16)a3.y; a[14] = (__bf16)a3.z; a[15] = (__bf16)a3.w;

        // B fragments: contiguous 32B per lane from the packed layout
        const size_t kt2 = ((size_t)(k0 >> 5) * 2 + half) * N;
        #pragma unroll
        for (int j = 0; j < NT; ++j) {
            v16bf bfr = *(const v16bf*)(Bp + (kt2 + tn + j * 16 + n16) * 16);
            c[j] = __builtin_amdgcn_wmma_f32_16x16x32_bf16(
                       /*neg_a=*/false, a, /*neg_b=*/false, bfr,
                       /*c_mod=*/(short)0, c[j],
                       /*reuse_a=*/false, /*reuse_b=*/false);
        }
    }

    #pragma unroll
    for (int j = 0; j < NT; ++j)
        #pragma unroll
        for (int r = 0; r < 8; ++r)
            C[(size_t)(tm + r + 8 * half) * ldc + tn + j * 16 + n16] = c[j][r];
}

// ---------------------------------------------------------------------------
// Causal depthwise conv1d (k=4, left pad 3) + bias + SiLU.
// Input: xz[:, :, 0:512] (row stride 1024). Output xi (row stride 512).
// ---------------------------------------------------------------------------
__global__ void conv_silu_kernel(const float* __restrict__ xz,
                                 const float* __restrict__ conv_w,
                                 const float* __restrict__ conv_b,
                                 float* __restrict__ xi) {
    int i = blockIdx.x * blockDim.x + threadIdx.x;
    if (i >= MROWS * D_INNER) return;
    int d = i & (D_INNER - 1);
    int r = i >> 9;              // b*T + t
    int t = r & (TT - 1);
    float acc = conv_b[d];
    #pragma unroll
    for (int j = 0; j < D_CONV; ++j) {
        int tt = t - (D_CONV - 1) + j;
        if (tt >= 0)
            acc = fmaf(xz[(size_t)(r - (D_CONV - 1) + j) * (2 * D_INNER) + d],
                       conv_w[d * D_CONV + j], acc);
    }
    xi[i] = silu_f(acc);
}

// ---------------------------------------------------------------------------
// dt = softplus(xdbl[:, 0:16] @ W_dt + b_dt)        (K = 16)
// ---------------------------------------------------------------------------
__global__ void dt_proj_kernel(const float* __restrict__ xdbl,
                               const float* __restrict__ W_dt,
                               const float* __restrict__ b_dt,
                               float* __restrict__ dtf) {
    int i = blockIdx.x * blockDim.x + threadIdx.x;
    if (i >= MROWS * D_INNER) return;
    int d = i & (D_INNER - 1);
    int r = i >> 9;
    const float* row = xdbl + (size_t)r * (DT_RANK + 2 * D_STATE);
    float s = b_dt[d];
    #pragma unroll
    for (int k = 0; k < DT_RANK; ++k)
        s = fmaf(row[k], W_dt[(size_t)k * D_INNER + d], s);
    dtf[i] = (s > 20.0f) ? s : log1pf(expf(s));
}

// ---------------------------------------------------------------------------
// Selective-scan + D skip + SiLU(z) gate, fused.
// One wave32 = two d-channels; lane (n = lane&15) owns state index n of
// channel half = lane>>4.  y = sum_n h*C via 4-step shfl_xor reduction.
// block = 256 threads = 16 channels; grid = B*D_INNER/16 = 64 blocks.
// ---------------------------------------------------------------------------
__global__ void scan_kernel(const float* __restrict__ xi,
                            const float* __restrict__ dtf,
                            const float* __restrict__ xdbl,
                            const float* __restrict__ xz,
                            const float* __restrict__ A_log,
                            const float* __restrict__ Dp,
                            float* __restrict__ yout) {
    const int tid  = threadIdx.x;
    const int lane = tid & 31;
    const int n    = lane & 15;
    const int half = lane >> 4;
    const int chl  = (tid >> 5) * 2 + half;
    const int g    = blockIdx.x * 16 + chl;   // 0 .. B*D_INNER-1
    const int b    = g >> 9;
    const int d    = g & (D_INNER - 1);

    const float A_dn = -expf(A_log[d * D_STATE + n]);
    const float D_d  = Dp[d];

    float h = 0.0f;
    const size_t rbase = (size_t)b * TT;
    for (int t = 0; t < TT; ++t) {
        size_t r  = rbase + t;
        float dt  = dtf[r * D_INNER + d];
        float xiv = xi [r * D_INNER + d];
        const float* xr = xdbl + r * (DT_RANK + 2 * D_STATE);
        float Bm = xr[DT_RANK + n];
        float Cm = xr[DT_RANK + D_STATE + n];

        float dA = expf(dt * A_dn);
        h = fmaf(dA, h, dt * Bm * xiv);

        float p = h * Cm;
        #pragma unroll
        for (int off = 1; off < D_STATE; off <<= 1)
            p += __shfl_xor(p, off, 32);       // stays within 16-lane group

        if (n == 0) {
            float y = p + xiv * D_d;
            float z = xz[r * (2 * D_INNER) + D_INNER + d];
            yout[r * D_INNER + d] = y * silu_f(z);
        }
    }
}

// ---------------------------------------------------------------------------
// Host-side launcher
// ---------------------------------------------------------------------------
extern "C" void kernel_launch(void* const* d_in, const int* in_sizes, int n_in,
                              void* d_out, int out_size, void* d_ws, size_t ws_size,
                              hipStream_t stream) {
    (void)in_sizes; (void)n_in; (void)out_size; (void)ws_size;

    const float* x = (const float*)d_in[0];
    // per-direction parameter pointers (fw at base 1, bw at base 10)
    const float* W_in[2]   = { (const float*)d_in[1],  (const float*)d_in[10] };
    const float* conv_w[2] = { (const float*)d_in[2],  (const float*)d_in[11] };
    const float* conv_b[2] = { (const float*)d_in[3],  (const float*)d_in[12] };
    const float* W_x[2]    = { (const float*)d_in[4],  (const float*)d_in[13] };
    const float* W_dt[2]   = { (const float*)d_in[5],  (const float*)d_in[14] };
    const float* b_dt[2]   = { (const float*)d_in[6],  (const float*)d_in[15] };
    const float* A_log[2]  = { (const float*)d_in[7],  (const float*)d_in[16] };
    const float* Dpar[2]   = { (const float*)d_in[8],  (const float*)d_in[17] };
    const float* W_out[2]  = { (const float*)d_in[9],  (const float*)d_in[18] };
    const float* merge_W   = (const float*)d_in[19];
    float* out = (float*)d_out;

    // ---- workspace layout (bytes) ----
    constexpr size_t SZ_XZ    = (size_t)MROWS * 2 * D_INNER * 4;              // 33.5 MB
    constexpr size_t SZ_XI    = (size_t)MROWS * D_INNER * 4;                  // 16.8 MB
    constexpr size_t SZ_XD    = (size_t)MROWS * (DT_RANK + 2 * D_STATE) * 4;  // 1.57 MB
    constexpr size_t SZ_DT    = SZ_XI;
    constexpr size_t SZ_Y     = SZ_XI;
    constexpr size_t SZ_WEFF  = (size_t)D_INNER * D_MODEL * 4;
    constexpr size_t SZ_WINB  = (size_t)D_MODEL * 2 * D_INNER * 2;
    constexpr size_t SZ_WXB   = (size_t)D_INNER * (DT_RANK + 2 * D_STATE) * 2;
    constexpr size_t SZ_WEFFB = (size_t)D_INNER * D_MODEL * 2;

    char* w = (char*)d_ws;
    size_t o = 0;
    float*  xz[2];    xz[0]    = (float*)(w + o); o += SZ_XZ;   xz[1]    = (float*)(w + o); o += SZ_XZ;
    float*  xi[2];    xi[0]    = (float*)(w + o); o += SZ_XI;   xi[1]    = (float*)(w + o); o += SZ_XI;
    float*  xd[2];    xd[0]    = (float*)(w + o); o += SZ_XD;   xd[1]    = (float*)(w + o); o += SZ_XD;
    float*  dtf[2];   dtf[0]   = (float*)(w + o); o += SZ_DT;   dtf[1]   = (float*)(w + o); o += SZ_DT;
    float*  yv[2];    yv[0]    = (float*)(w + o); o += SZ_Y;    yv[1]    = (float*)(w + o); o += SZ_Y;
    float*  weff[2];  weff[0]  = (float*)(w + o); o += SZ_WEFF; weff[1]  = (float*)(w + o); o += SZ_WEFF;
    __bf16* winb[2];  winb[0]  = (__bf16*)(w + o); o += SZ_WINB; winb[1] = (__bf16*)(w + o); o += SZ_WINB;
    __bf16* wxb[2];   wxb[0]   = (__bf16*)(w + o); o += SZ_WXB;  wxb[1]  = (__bf16*)(w + o); o += SZ_WXB;
    __bf16* weffb[2]; weffb[0] = (__bf16*)(w + o); o += SZ_WEFFB; weffb[1] = (__bf16*)(w + o); o += SZ_WEFFB;

    const int TPB = 256;
    const int N_ED = MROWS * D_INNER;   // elementwise grid size
    const int NX   = DT_RANK + 2 * D_STATE;   // 48

    // ---- 1. fuse W_out with merge halves; pack weights to bf16 B-frags ----
    for (int dir = 0; dir < 2; ++dir) {
        fuse_w_kernel<<<(D_INNER * D_MODEL + TPB - 1) / TPB, TPB, 0, stream>>>(
            W_out[dir], merge_W + (size_t)dir * D_MODEL * D_MODEL, weff[dir]);
    }
    for (int dir = 0; dir < 2; ++dir) {
        int nwin = D_MODEL * 2 * D_INNER;
        int nwx  = D_INNER * NX;
        int nwe  = D_INNER * D_MODEL;
        pack_b_kernel<<<(nwin + TPB - 1) / TPB, TPB, 0, stream>>>(
            W_in[dir], winb[dir], D_MODEL, 2 * D_INNER);
        pack_b_kernel<<<(nwx + TPB - 1) / TPB, TPB, 0, stream>>>(
            W_x[dir], wxb[dir], D_INNER, NX);
        pack_b_kernel<<<(nwe + TPB - 1) / TPB, TPB, 0, stream>>>(
            weff[dir], weffb[dir], D_INNER, D_MODEL);
    }

    // ---- 2. in-projection GEMM (bw direction reads time-flipped x) ----
    for (int dir = 0; dir < 2; ++dir) {
        dim3 g(2 * D_INNER / 64, MROWS / 16);
        wmma_gemm_kernel<4><<<g, 32, 0, stream>>>(
            x, winb[dir], xz[dir],
            MROWS, 2 * D_INNER, D_MODEL,
            D_MODEL, 2 * D_INNER,
            /*flipT=*/dir ? TT : 0, /*accum=*/0);
    }

    // ---- 3. causal depthwise conv + SiLU ----
    for (int dir = 0; dir < 2; ++dir)
        conv_silu_kernel<<<(N_ED + TPB - 1) / TPB, TPB, 0, stream>>>(
            xz[dir], conv_w[dir], conv_b[dir], xi[dir]);

    // ---- 4. x-projection GEMM: xdbl = xi @ W_x   (N = 48 -> NT = 3) ----
    for (int dir = 0; dir < 2; ++dir) {
        dim3 g(1, MROWS / 16);
        wmma_gemm_kernel<3><<<g, 32, 0, stream>>>(
            xi[dir], wxb[dir], xd[dir],
            MROWS, NX, D_INNER,
            D_INNER, NX,
            /*flipT=*/0, /*accum=*/0);
    }

    // ---- 5. dt projection + softplus ----
    for (int dir = 0; dir < 2; ++dir)
        dt_proj_kernel<<<(N_ED + TPB - 1) / TPB, TPB, 0, stream>>>(
            xd[dir], W_dt[dir], b_dt[dir], dtf[dir]);

    // ---- 6. fused selective scan + skip + gate ----
    for (int dir = 0; dir < 2; ++dir)
        scan_kernel<<<BB * D_INNER / 16, 256, 0, stream>>>(
            xi[dir], dtf[dir], xd[dir], xz[dir], A_log[dir], Dpar[dir], yv[dir]);

    // ---- 7. fused out-projection + merge:
    //      out = y_f @ W_eff_f + flip(y_b) @ W_eff_b ----
    {
        dim3 g(D_MODEL / 64, MROWS / 16);
        wmma_gemm_kernel<4><<<g, 32, 0, stream>>>(
            yv[0], weffb[0], out,
            MROWS, D_MODEL, D_INNER,
            D_INNER, D_MODEL, /*flipT=*/0, /*accum=*/0);
        wmma_gemm_kernel<4><<<g, 32, 0, stream>>>(
            yv[1], weffb[1], out,
            MROWS, D_MODEL, D_INNER,
            D_INNER, D_MODEL, /*flipT=*/TT, /*accum=*/1);
    }
}
